// slotGAT_22402549416539
// MI455X (gfx1250) — matrix operand order repaired and checked by reading, not compiled, not run on hardware
//
#include <hip/hip_runtime.h>
#include <math.h>

// ---- problem constants (match reference) ----
#define TT    3
#define NPER  20000
#define NN    60000      // TT*NPER
#define EE    480000
#define BBB   200000
#define RR    5
#define NEGS  0.2f
#define ALPHA 0.05f

typedef __attribute__((ext_vector_type(16))) _Float16 v16h;
typedef __attribute__((ext_vector_type(8)))  float    v8f;

// ---- CDNA5 async global->LDS path (guarded: falls back to float4 staging) ----
#if defined(__has_builtin)
#  if __has_builtin(__builtin_amdgcn_global_load_async_to_lds_b128)
#    define HAVE_ASYNC_LDS 1
#  endif
#endif
#ifndef HAVE_ASYNC_LDS
#  define HAVE_ASYNC_LDS 0
#endif

typedef int v4i_ __attribute__((vector_size(16)));   // gcc-style, matches builtin proto

__device__ __forceinline__ void stage16(const float* g, float* l) {
#if HAVE_ASYNC_LDS
  typedef __attribute__((address_space(1))) v4i_ GT;   // global (prints as __device__)
  typedef __attribute__((address_space(3))) v4i_ LT;   // LDS
  // generic LDS pointer low 32 bits == LDS offset (ISA aperture rule)
  __builtin_amdgcn_global_load_async_to_lds_b128(
      (GT*)(size_t)g, (LT*)(unsigned)(size_t)l, 0, 0);
#else
  *(float4*)l = *(const float4*)g;
#endif
}

__device__ __forceinline__ void stage_fence_pre() {
#if HAVE_ASYNC_LDS
  // async engine writes are unordered vs this wave's ds reads: drain ds first
  asm volatile("s_wait_dscnt 0x0" ::: "memory");
#endif
}

__device__ __forceinline__ void stage_fence_post() {
#if HAVE_ASYNC_LDS
#  if __has_builtin(__builtin_amdgcn_s_wait_asynccnt)
  __builtin_amdgcn_s_wait_asynccnt(0);
#  else
  asm volatile("s_wait_asynccnt 0x0" ::: "memory");
#  endif
#endif
}

// -------------------------------------------------------------------------
// zero fill
__global__ void fill_zero(float* p, long n) {
  long i = (long)blockIdx.x * blockDim.x + threadIdx.x;
  if (i < n) p[i] = 0.0f;
}

// -------------------------------------------------------------------------
// WMMA GEMM: C[M x Nc] = A[M x K] * W[K x Nc]
//   grid = (M/16, Nc/32), block = 32: one wave -> 16x32 output (2 WMMAs/K-step)
//   A,W staged through LDS with b128 granularity; K % 32 == 0.
//   LDS pitch 44 floats: 16B-aligned rows, 16 distinct banks across l16 rows.
#define LAS 44

__global__ void wmma_gemm(const float* __restrict__ A, int lda,
                          const float* __restrict__ W, int ldw,
                          float* __restrict__ C, int ldc, int K) {
  const int mt   = blockIdx.x * 16;
  const int nt   = blockIdx.y * 32;
  const int lane = threadIdx.x;
  const int half = lane >> 4;
  const int l16  = lane & 15;
  __shared__ float lA[16 * LAS];   // 16 rows x 32 cols (f32)
  __shared__ float lB[32 * LAS];   // 32 rows x 32 cols (f32)
  v8f acc0 = {}, acc1 = {};
  for (int k0 = 0; k0 < K; k0 += 32) {
    stage_fence_pre();
#pragma unroll
    for (int it = 0; it < 4; ++it) {             // A tile: 128 float4 chunks
      int c = it * 32 + lane, r = c >> 3, c4 = (c & 7) << 2;
      stage16(A + (size_t)(mt + r) * lda + (k0 + c4), &lA[r * LAS + c4]);
    }
#pragma unroll
    for (int it = 0; it < 8; ++it) {             // B tile: 256 float4 chunks
      int c = it * 32 + lane, r = c >> 3, c4 = (c & 7) << 2;
      stage16(W + (size_t)(k0 + r) * ldw + (nt + c4), &lB[r * LAS + c4]);
    }
    stage_fence_post();
    __syncthreads();
    // A fragment (ISA 16-bit A 16x32 layout): row=l16,
    //   i<8 : K = half*8 + i ; i>=8 : K = 16 + half*8 + (i-8)
    const float* ar = &lA[l16 * LAS + (half << 3)];
    float4 a0 = *(const float4*)(ar);
    float4 a1 = *(const float4*)(ar + 4);
    float4 a2 = *(const float4*)(ar + 16);
    float4 a3 = *(const float4*)(ar + 20);
    v16h av;
    av[0]=(_Float16)a0.x; av[1]=(_Float16)a0.y; av[2]=(_Float16)a0.z; av[3]=(_Float16)a0.w;
    av[4]=(_Float16)a1.x; av[5]=(_Float16)a1.y; av[6]=(_Float16)a1.z; av[7]=(_Float16)a1.w;
    av[8]=(_Float16)a2.x; av[9]=(_Float16)a2.y; av[10]=(_Float16)a2.z; av[11]=(_Float16)a2.w;
    av[12]=(_Float16)a3.x; av[13]=(_Float16)a3.y; av[14]=(_Float16)a3.z; av[15]=(_Float16)a3.w;
    // B fragments (16-bit B 32x16 layout): col = l16, i<8: K = half*8 + i ; i>=8: K = 16 + half*8 + (i-8)
#pragma unroll
    for (int ns = 0; ns < 2; ++ns) {
      v16h bv;
#pragma unroll
      for (int i = 0; i < 16; ++i)
        bv[i] = (_Float16)lB[(size_t)((half << 3) + i + ((i >> 3) << 3)) * LAS + (ns << 4) + l16];
      if (ns == 0)
        acc0 = __builtin_amdgcn_wmma_f32_16x16x32_f16(false, av, false, bv,
                                                      (short)0, acc0, false, false);
      else
        acc1 = __builtin_amdgcn_wmma_f32_16x16x32_f16(false, av, false, bv,
                                                      (short)0, acc1, false, false);
    }
    __syncthreads();
  }
#pragma unroll
  for (int j = 0; j < 8; ++j) {
    size_t row = (size_t)(mt + j + (half << 3));
    C[row * ldc + (nt + l16)]      = acc0[j];
    C[row * ldc + (nt + 16 + l16)] = acc1[j];
  }
}

// -------------------------------------------------------------------------
// h0 bias add (only the node's own slot is nonzero) + emb0 column of z
__global__ void emb0_kernel(float* __restrict__ h0, const float* __restrict__ fc_b,
                            float* __restrict__ z) {
  int n = blockIdx.x, k = threadIdx.x;            // 96 threads
  int tn = n / NPER;
  float v = h0[(size_t)n * 96 + k];
  if ((k >> 5) == tn) { v += fc_b[tn * 32 + (k & 31)]; h0[(size_t)n * 96 + k] = v; }
  __shared__ float g[96];
  __shared__ float nrm;
  g[k] = v;
  __syncthreads();
  if (k == 0) { float s = 0.f; for (int i = 0; i < 96; ++i) s += g[i] * g[i]; nrm = sqrtf(s); }
  __syncthreads();
  if (k < 32) {
    float inv = 1.0f / fmaxf(nrm, 1e-12f);
    z[(size_t)n * 128 + k] = (g[k] + g[32 + k] + g[64 + k]) * inv * (1.0f / 3.0f);
  }
}

// -------------------------------------------------------------------------
// edge-type logits: elog[r,h] = dot((eemb[r]@fce_w + fce_b)[h*64:(h+1)*64], ae[h])
__global__ void edge_logit_kernel(const float* __restrict__ eemb,
                                  const float* __restrict__ fw,
                                  const float* __restrict__ fb,
                                  const float* __restrict__ ae,
                                  float* __restrict__ elog, int heads) {
  int i = threadIdx.x;
  if (i >= RR * heads) return;
  int r = i / heads, h = i - r * heads;
  float acc = 0.f;
  for (int d = 0; d < 64; ++d) {
    float ef = fb[h * 64 + d];
    for (int j = 0; j < 64; ++j)
      ef += eemb[r * 64 + j] * fw[j * (heads * 64) + h * 64 + d];
    acc += ef * ae[h * 64 + d];
  }
  elog[r * heads + h] = acc;
}

// -------------------------------------------------------------------------
// f[n,h,t*32+d] = ftmp[n,t,h*32+d]   (ftmp row = T * (heads*32))
__global__ void make_f(const float* __restrict__ ftmp, int outps,
                       float* __restrict__ f, int heads) {
  int n = blockIdx.x, tid = threadIdx.x;          // heads*96 threads
  int h = tid / 96, k = tid - h * 96;
  int t = k >> 5, d = k & 31;
  f[((size_t)n * heads + h) * 96 + k] = ftmp[((size_t)n * TT + t) * outps + h * 32 + d];
}

// el/er per (node, head): 96-length dots vs al/ar
__global__ void node_logits(const float* __restrict__ f, const float* __restrict__ al,
                            const float* __restrict__ ar, float* __restrict__ el,
                            float* __restrict__ er, int heads) {
  int i = blockIdx.x * blockDim.x + threadIdx.x;
  if (i >= NN * heads) return;
  int h = i % heads;
  const float* fr = f + (size_t)i * 96;
  float sl = 0.f, sr = 0.f;
  for (int k = 0; k < 96; ++k) { float v = fr[k]; sl += v * al[h * 96 + k]; sr += v * ar[h * 96 + k]; }
  el[i] = sl; er[i] = sr;
}

// -------------------------------------------------------------------------
__device__ __forceinline__ unsigned okey(float f) {
  unsigned u = __float_as_uint(f);
  return (u & 0x80000000u) ? ~u : (u | 0x80000000u);
}
__device__ __forceinline__ float okey_dec(unsigned k) {
  return (k & 0x80000000u) ? __uint_as_float(k & 0x7FFFFFFFu) : __uint_as_float(~k);
}
__device__ __forceinline__ float edge_e(const int* src, const int* dst, const int* ety,
                                        const float* el, const float* er,
                                        const float* elog, int e, int h, int heads) {
  float ev = el[src[e] * heads + h] + er[dst[e] * heads + h] + elog[ety[e] * heads + h];
  return ev > 0.f ? ev : NEGS * ev;               // leaky_relu
}

__global__ void edge_max(const int* __restrict__ src, const int* __restrict__ dst,
                         const int* __restrict__ ety, const float* __restrict__ el,
                         const float* __restrict__ er, const float* __restrict__ elog,
                         unsigned* __restrict__ mkey, int heads) {
  int i = blockIdx.x * blockDim.x + threadIdx.x;
  if (i >= EE * heads) return;
  int e = i / heads, h = i - e * heads;
  float ev = edge_e(src, dst, ety, el, er, elog, e, h, heads);
  atomicMax(&mkey[dst[e] * heads + h], okey(ev));
}

__global__ void edge_exp(const int* __restrict__ src, const int* __restrict__ dst,
                         const int* __restrict__ ety, const float* __restrict__ el,
                         const float* __restrict__ er, const float* __restrict__ elog,
                         const unsigned* __restrict__ mkey, float* __restrict__ ssum,
                         float* __restrict__ exbuf, int heads) {
  int i = blockIdx.x * blockDim.x + threadIdx.x;
  if (i >= EE * heads) return;
  int e = i / heads, h = i - e * heads;
  float ev = edge_e(src, dst, ety, el, er, elog, e, h, heads);
  float m  = okey_dec(mkey[dst[e] * heads + h]);
  float ex = expf(ev - m);
  exbuf[i] = ex;
  atomicAdd(&ssum[dst[e] * heads + h], ex);
}

__global__ void edge_norm(const float* __restrict__ exbuf, const float* __restrict__ ssum,
                          const int* __restrict__ dst, const float* __restrict__ resattn,
                          float* __restrict__ aout, int heads) {
  int i = blockIdx.x * blockDim.x + threadIdx.x;
  if (i >= EE * heads) return;
  int e = i / heads, h = i - e * heads;
  float a = exbuf[i] / (ssum[dst[e] * heads + h] + 1e-9f);
  if (resattn) a = a * (1.0f - ALPHA) + resattn[i] * ALPHA;
  aout[i] = a;
}

// rst[dst,h,:] += a[e,h] * f[src,h,:]    grid = EE, block = heads*96
__global__ void scatter_rst(const int* __restrict__ src, const int* __restrict__ dst,
                            const float* __restrict__ a, const float* __restrict__ f,
                            float* __restrict__ rst, int heads) {
  int e = blockIdx.x, tid = threadIdx.x;
  int h = tid / 96, k = tid - h * 96;
  float av = a[(size_t)e * heads + h];
  int sN = src[e], dN = dst[e];
  atomicAdd(&rst[((size_t)dN * heads + h) * 96 + k],
            av * f[((size_t)sN * heads + h) * 96 + k]);
}

// -------------------------------------------------------------------------
// layers 0/1 epilogue: (+residual)+bias, ELU, write h_next [N,T,64], emb -> z col
__global__ void post01(const float* __restrict__ rst, const float* __restrict__ bias,
                       const float* __restrict__ hin, float* __restrict__ hout,
                       float* __restrict__ z, int zcol) {
  int n = blockIdx.x, tid = threadIdx.x;          // 192 threads (heads=2)
  int h = tid / 96, k = tid - h * 96;
  int t = k >> 5, d = k & 31;
  float v = rst[((size_t)n * 2 + h) * 96 + k] + bias[h * 96 + k];
  if (hin) v += hin[((size_t)n * TT + t) * 64 + h * 32 + d];
  v = v > 0.f ? v : expm1f(v);                    // ELU
  hout[((size_t)n * TT + t) * 64 + h * 32 + d] = v;
  __shared__ float sf[192];
  __shared__ float g[96];
  __shared__ float nrm;
  sf[tid] = v;
  __syncthreads();
  if (h == 0) g[k] = 0.5f * (sf[k] + sf[96 + k]); // mean over heads
  __syncthreads();
  if (tid == 0) { float s = 0.f; for (int i = 0; i < 96; ++i) s += g[i] * g[i]; nrm = sqrtf(s); }
  __syncthreads();
  if (tid < 32) {
    float inv = 1.0f / fmaxf(nrm, 1e-12f);
    z[(size_t)n * 128 + zcol + tid] = (g[tid] + g[32 + tid] + g[64 + tid]) * inv * (1.0f / 3.0f);
  }
}

// layer 2 epilogue (heads=1, residual via hres [N,T,32], no activation) -> z col 96
__global__ void post2(const float* __restrict__ rst, const float* __restrict__ bias,
                      const float* __restrict__ hres, float* __restrict__ z) {
  int n = blockIdx.x, k = threadIdx.x;            // 96 threads
  int t = k >> 5, d = k & 31;
  float v = rst[(size_t)n * 96 + k] + bias[k] + hres[((size_t)n * TT + t) * 32 + d];
  __shared__ float g[96];
  __shared__ float nrm;
  g[k] = v;
  __syncthreads();
  if (k == 0) { float s = 0.f; for (int i = 0; i < 96; ++i) s += g[i] * g[i]; nrm = sqrtf(s); }
  __syncthreads();
  if (k < 32) {
    float inv = 1.0f / fmaxf(nrm, 1e-12f);
    z[(size_t)n * 128 + 96 + k] = (g[k] + g[32 + k] + g[64 + k]) * inv * (1.0f / 3.0f);
  }
}

// -------------------------------------------------------------------------
// score[b] = dot(U_r[left[b]], z[right[b]]) for b with mid[b]==r; one wave per b
__global__ void score_r(const float* __restrict__ U, const float* __restrict__ z,
                        const int* __restrict__ left, const int* __restrict__ right,
                        const int* __restrict__ mid, float* __restrict__ out, int r) {
  int gid = blockIdx.x * blockDim.x + threadIdx.x;
  int b = gid >> 5, lane = gid & 31;
  if (b >= BBB) return;
  if (mid[b] != r) return;
  const float4 u = ((const float4*)(U + (size_t)left[b]  * 128))[lane];
  const float4 v = ((const float4*)(z + (size_t)right[b] * 128))[lane];
  float p = u.x * v.x + u.y * v.y + u.z * v.z + u.w * v.w;
#pragma unroll
  for (int o = 16; o; o >>= 1) p += __shfl_down(p, o, 32);
  if (lane == 0) out[b] = p;
}

// -------------------------------------------------------------------------
extern "C" void kernel_launch(void* const* d_in, const int* in_sizes, int n_in,
                              void* d_out, int out_size, void* d_ws, size_t ws_size,
                              hipStream_t stream) {
  (void)in_sizes; (void)n_in; (void)out_size; (void)ws_size;
  const float* feats  = (const float*)d_in[0];
  const float* fc_w   = (const float*)d_in[1];
  const float* fc_b   = (const float*)d_in[2];
  const float* w0     = (const float*)d_in[3];
  const float* al0    = (const float*)d_in[4];
  const float* ar0    = (const float*)d_in[5];
  const float* b0     = (const float*)d_in[6];
  const float* eemb0  = (const float*)d_in[7];
  const float* fcew0  = (const float*)d_in[8];
  const float* fceb0  = (const float*)d_in[9];
  const float* ae0    = (const float*)d_in[10];
  const float* w1     = (const float*)d_in[11];
  const float* al1    = (const float*)d_in[12];
  const float* ar1    = (const float*)d_in[13];
  const float* b1     = (const float*)d_in[14];
  const float* eemb1  = (const float*)d_in[15];
  const float* fcew1  = (const float*)d_in[16];
  const float* fceb1  = (const float*)d_in[17];
  const float* ae1    = (const float*)d_in[18];
  const float* w2     = (const float*)d_in[19];
  const float* al2    = (const float*)d_in[20];
  const float* ar2    = (const float*)d_in[21];
  const float* b2     = (const float*)d_in[22];
  const float* eemb2  = (const float*)d_in[23];
  const float* fcew2  = (const float*)d_in[24];
  const float* fceb2  = (const float*)d_in[25];
  const float* ae2    = (const float*)d_in[26];
  const float* resw2  = (const float*)d_in[27];
  const float* Wdm    = (const float*)d_in[28];
  const int*   src    = (const int*)d_in[29];
  const int*   dst    = (const int*)d_in[30];
  const int*   ety    = (const int*)d_in[31];
  const int*   left   = (const int*)d_in[32];
  const int*   right  = (const int*)d_in[33];
  const int*   mid    = (const int*)d_in[34];
  float* out = (float*)d_out;

  // ---- workspace layout (floats) : ~81.2M floats (~325 MB) ----
  float* ws   = (float*)d_ws;
  float* h0   = ws;                    // N*96   (reused as hres later)
  float* hA   = h0   + (size_t)NN * 96;
  float* hB   = hA   + (size_t)NN * 192;
  float* ftmp = hB   + (size_t)NN * 192;
  float* fbuf = ftmp + (size_t)NN * 192;
  float* rst  = fbuf + (size_t)NN * 192;
  float* zb   = rst  + (size_t)NN * 192;   // N*128
  float* Ub   = zb   + (size_t)NN * 128;   // N*128 (one relation at a time)
  float* el   = Ub   + (size_t)NN * 128;   // N*2
  float* er   = el   + (size_t)NN * 2;
  float* mk   = er   + (size_t)NN * 2;     // as unsigned
  float* ss   = mk   + (size_t)NN * 2;
  float* abuf = ss   + (size_t)NN * 2;     // E*2
  float* aprv = abuf + (size_t)EE * 2;
  float* elg0 = aprv + (size_t)EE * 2;     // tiny
  float* elg1 = elg0 + 16;
  float* elg2 = elg1 + 16;

  auto zero = [&](float* p, long n) {
    fill_zero<<<dim3((unsigned)((n + 255) / 256)), dim3(256), 0, stream>>>(p, n);
  };
  auto gemm = [&](const float* A, int lda, const float* W, int ldw,
                  float* C, int ldc, int M, int Nc, int K) {
    wmma_gemm<<<dim3(M / 16, Nc / 32), dim3(32), 0, stream>>>(A, lda, W, ldw, C, ldc, K);
  };
  const int EH2 = (EE * 2 + 255) / 256, EH1 = (EE + 255) / 256;

  // ---------- input FC -> h0 slots + emb0 ----------
  zero(h0, (long)NN * 96);
  for (int t = 0; t < TT; ++t)
    gemm(feats + (size_t)t * NPER * 64, 64, fc_w + (size_t)t * 64 * 32, 32,
         h0 + (size_t)t * NPER * 96 + t * 32, 96, NPER, 32, 64);
  emb0_kernel<<<NN, 96, 0, stream>>>(h0, fc_b, zb);

  // ---------- layer 0 ----------
  edge_logit_kernel<<<1, 32, 0, stream>>>(eemb0, fcew0, fceb0, ae0, elg0, 2);
  for (int t = 0; t < TT; ++t)
    gemm(h0 + t * 32, 96, w0 + (size_t)t * 32 * 64, 64, ftmp + t * 64, 192, NN, 64, 32);
  make_f<<<NN, 192, 0, stream>>>(ftmp, 64, fbuf, 2);
  node_logits<<<(NN * 2 + 127) / 128, 128, 0, stream>>>(fbuf, al0, ar0, el, er, 2);
  zero(mk, NN * 2); zero(ss, NN * 2);
  edge_max<<<EH2, 256, 0, stream>>>(src, dst, ety, el, er, elg0, (unsigned*)mk, 2);
  edge_exp<<<EH2, 256, 0, stream>>>(src, dst, ety, el, er, elg0, (const unsigned*)mk, ss, abuf, 2);
  edge_norm<<<EH2, 256, 0, stream>>>(abuf, ss, dst, nullptr, aprv, 2);
  zero(rst, (long)NN * 192);
  scatter_rst<<<EE, 192, 0, stream>>>(src, dst, aprv, fbuf, rst, 2);
  post01<<<NN, 192, 0, stream>>>(rst, b0, nullptr, hA, zb, 32);

  // ---------- layer 1 (residual = input, res_attn = aprv) ----------
  edge_logit_kernel<<<1, 32, 0, stream>>>(eemb1, fcew1, fceb1, ae1, elg1, 2);
  for (int t = 0; t < TT; ++t)
    gemm(hA + t * 64, 192, w1 + (size_t)t * 64 * 64, 64, ftmp + t * 64, 192, NN, 64, 64);
  make_f<<<NN, 192, 0, stream>>>(ftmp, 64, fbuf, 2);
  node_logits<<<(NN * 2 + 127) / 128, 128, 0, stream>>>(fbuf, al1, ar1, el, er, 2);
  zero(mk, NN * 2); zero(ss, NN * 2);
  edge_max<<<EH2, 256, 0, stream>>>(src, dst, ety, el, er, elg1, (unsigned*)mk, 2);
  edge_exp<<<EH2, 256, 0, stream>>>(src, dst, ety, el, er, elg1, (const unsigned*)mk, ss, abuf, 2);
  edge_norm<<<EH2, 256, 0, stream>>>(abuf, ss, dst, aprv, abuf, 2);
  zero(rst, (long)NN * 192);
  scatter_rst<<<EE, 192, 0, stream>>>(src, dst, abuf, fbuf, rst, 2);
  post01<<<NN, 192, 0, stream>>>(rst, b1, hA, hB, zb, 64);

  // ---------- layer 2 (heads=1, residual via res_w2, no act) ----------
  edge_logit_kernel<<<1, 32, 0, stream>>>(eemb2, fcew2, fceb2, ae2, elg2, 1);
  for (int t = 0; t < TT; ++t)
    gemm(hB + t * 64, 192, w2 + (size_t)t * 64 * 32, 32, ftmp + t * 32, 96, NN, 32, 64);
  make_f<<<NN, 96, 0, stream>>>(ftmp, 32, fbuf, 1);
  node_logits<<<(NN + 127) / 128, 128, 0, stream>>>(fbuf, al2, ar2, el, er, 1);
  zero(mk, NN); zero(ss, NN);
  edge_max<<<EH1, 256, 0, stream>>>(src, dst, ety, el, er, elg2, (unsigned*)mk, 1);
  edge_exp<<<EH1, 256, 0, stream>>>(src, dst, ety, el, er, elg2, (const unsigned*)mk, ss, abuf, 1);
  edge_norm<<<EH1, 256, 0, stream>>>(abuf, ss, dst, nullptr, abuf, 1);
  zero(rst, (long)NN * 96);
  scatter_rst<<<EE, 96, 0, stream>>>(src, dst, abuf, fbuf, rst, 1);
  // residual: hres = (hB as [(N*T) x 64]) @ res_w2 -> reuse h0 buffer [(N*T) x 32]
  gemm(hB, 64, resw2, 32, h0, 32, NN * TT, 32, 64);
  post2<<<NN, 96, 0, stream>>>(rst, b2, h0, zb);

  // ---------- DistMult: per relation, U = Z @ Wdm[r], then gather-dot ----------
  for (int r = 0; r < RR; ++r) {
    gemm(zb, 128, Wdm + (size_t)r * 128 * 128, 128, Ub, 128, NN, 128, 128);
    score_r<<<(BBB * 32 + 255) / 256, 256, 0, stream>>>(Ub, zb, left, right, mid, out, r);
  }
}
